// MoeLayer_10307921510767
// MI455X (gfx1250) — matrix operand order repaired and checked by reading, not compiled, not run on hardware
//
#include <hip/hip_runtime.h>
#include <math.h>
#include <stdint.h>

// Problem constants (from reference)
#define T_TOK 8192      // B*S = 4*2048
#define DD    1024      // embedding dim
#define HH    2048      // hidden dim
#define EE    8         // experts
#define NSEG  16        // (expert, k) segments
#define NT    160       // max M-tiles: 2T/128 + 32 slack
#define BM    128
#define BN    64
#define BK    32
#define LDA   40        // padded LDS row stride (bf16 elems)

typedef __attribute__((ext_vector_type(16))) __bf16 v16bf;
typedef __attribute__((ext_vector_type(8)))  __bf16 v8bf;
typedef __attribute__((ext_vector_type(4)))  __bf16 v4bf;
typedef __attribute__((ext_vector_type(8)))  float  v8f;

// ---------------------------------------------------------------------------
// CDNA5 async global->LDS copy (no VGPR round trip, tracked by ASYNCcnt)
// ---------------------------------------------------------------------------
__device__ inline void async_ld16(unsigned ldsOffBytes, const __bf16* g)
{
    asm volatile("global_load_async_to_lds_b128 %0, %1, off"
                 :: "v"(ldsOffBytes),
                    "v"((unsigned long long)(uintptr_t)g)
                 : "memory");
}
__device__ inline void wait_async0()
{
    asm volatile("s_wait_asynccnt 0x0" ::: "memory");
}

// ---------------------------------------------------------------------------
// Pass 1: router (wave-per-token) + x -> bf16 conversion
// ---------------------------------------------------------------------------
__global__ __launch_bounds__(256) void moe_router(
    const float* __restrict__ x, const float* __restrict__ Wg,
    __bf16* __restrict__ xb, int* __restrict__ tokE,
    float* __restrict__ tokW, int* __restrict__ cnt)
{
    const int wave = threadIdx.x >> 5;
    const int lane = threadIdx.x & 31;
    const int t    = blockIdx.x * 8 + wave;
    const float* xr = x + (size_t)t * DD;

    float acc[EE];
#pragma unroll
    for (int e = 0; e < EE; ++e) acc[e] = 0.f;

#pragma unroll
    for (int i = 0; i < 8; ++i) {
        const int d4 = i * 32 + lane;            // float4 index within row
        float4 xv = ((const float4*)xr)[d4];
        v4bf bv;
        bv[0] = (__bf16)xv.x; bv[1] = (__bf16)xv.y;
        bv[2] = (__bf16)xv.z; bv[3] = (__bf16)xv.w;
        *(v4bf*)(xb + (size_t)t * DD + d4 * 4) = bv;
        float xs[4] = {xv.x, xv.y, xv.z, xv.w};
#pragma unroll
        for (int j = 0; j < 4; ++j) {
            const float* wr = Wg + (size_t)(d4 * 4 + j) * EE;
#pragma unroll
            for (int e = 0; e < EE; ++e) acc[e] += xs[j] * wr[e];
        }
    }
#pragma unroll
    for (int off = 16; off > 0; off >>= 1) {
#pragma unroll
        for (int e = 0; e < EE; ++e) acc[e] += __shfl_xor(acc[e], off, 32);
    }
    if (lane == 0) {
        int e0 = 0; float v0 = acc[0];
#pragma unroll
        for (int e = 1; e < EE; ++e) if (acc[e] > v0) { v0 = acc[e]; e0 = e; }
        int e1 = -1; float v1 = -3.4e38f;
#pragma unroll
        for (int e = 0; e < EE; ++e)
            if (e != e0 && acc[e] > v1) { v1 = acc[e]; e1 = e; }
        float p1 = __expf(v1 - v0);              // softmax over {v0, v1}, v0 >= v1
        float inv = __builtin_amdgcn_rcpf(1.f + p1);
        tokE[t * 2 + 0] = e0;  tokE[t * 2 + 1] = e1;
        tokW[t * 2 + 0] = inv;  tokW[t * 2 + 1] = p1 * inv;
        atomicAdd(&cnt[e0 * 2 + 0], 1);
        atomicAdd(&cnt[e1 * 2 + 1], 1);
    }
}

// ---------------------------------------------------------------------------
// Pass 2: prefix-sum segments + build flat M-tile table (single thread)
// ---------------------------------------------------------------------------
__global__ void moe_scan(const int* __restrict__ cnt, int* __restrict__ segOff,
                         int* __restrict__ fill, int* __restrict__ tileSeg,
                         int* __restrict__ tileSlot)
{
    if (threadIdx.x == 0 && blockIdx.x == 0) {
        int off = 0;
        for (int s = 0; s < NSEG; ++s) { segOff[s] = off; off += cnt[s]; fill[s] = 0; }
        segOff[NSEG] = off;
        int t = 0;
        for (int s = 0; s < NSEG; ++s) {
            const int c = cnt[s];
            for (int r = 0; r < c; r += BM) { tileSeg[t] = s; tileSlot[t] = segOff[s] + r; ++t; }
        }
        for (; t < NT; ++t) { tileSeg[t] = -1; tileSlot[t] = 0; }
    }
}

// ---------------------------------------------------------------------------
// Pass 3: scatter token ids into per-(expert,k) lists
// ---------------------------------------------------------------------------
__global__ __launch_bounds__(256) void moe_scatter(
    const int* __restrict__ tokE, int* __restrict__ fill,
    const int* __restrict__ segOff, int* __restrict__ entryTok)
{
    const int t = blockIdx.x * blockDim.x + threadIdx.x;
    if (t < T_TOK) {
#pragma unroll
        for (int k = 0; k < 2; ++k) {
            const int s = tokE[t * 2 + k] * 2 + k;
            const int pos = atomicAdd(&fill[s], 1);
            entryTok[segOff[s] + pos] = t;
        }
    }
}

// ---------------------------------------------------------------------------
// Fragment helpers (ISA 7.12.2 16-bit layouts, wave32)
//   A (16xK rows=M): lane L holds row L%16, K chunks [kb,kb+8) & [kb+16,kb+24), kb = L<16?0:8
//   B (Kx16 cols=N): lane L holds col L%16, K chunks [kb,kb+8) & [kb+16,kb+24), kb = L<16?0:8
// ---------------------------------------------------------------------------
__device__ inline v16bf load_fragA(const __bf16* lds, int row, int lane)
{
    const int kb = (lane < 16) ? 0 : 8;
    v8bf lo = *(const v8bf*)&lds[row * LDA + kb];
    v8bf hi = *(const v8bf*)&lds[row * LDA + kb + 16];
    v16bf a;
#pragma unroll
    for (int i = 0; i < 8; ++i) { a[i] = lo[i]; a[i + 8] = hi[i]; }
    return a;
}
__device__ inline v16bf load_fragB(const __bf16* lds, int col, int lane)
{
    const int kb = (lane < 16) ? 0 : 8;
    v8bf lo = *(const v8bf*)&lds[col * LDA + kb];
    v8bf hi = *(const v8bf*)&lds[col * LDA + kb + 16];
    v16bf b;
#pragma unroll
    for (int i = 0; i < 8; ++i) { b[i] = lo[i]; b[i + 8] = hi[i]; }
    return b;
}

// ---------------------------------------------------------------------------
// Pass 4: grouped GEMM  u = x W1[e], v = x W2[e]  (gathered rows) + SwiGLU
//   double-buffered LDS software pipeline; A tiles via async global->LDS DMA
// ---------------------------------------------------------------------------
__global__ __launch_bounds__(256) void moe_gemm_uv(
    const __bf16* __restrict__ xb, const float* __restrict__ W1,
    const float* __restrict__ W2, __bf16* __restrict__ hb,
    const int* __restrict__ entryTok, const int* __restrict__ tileSeg,
    const int* __restrict__ tileSlot, const int* __restrict__ segOff)
{
    __shared__ __bf16 Al[2][BM * LDA];
    __shared__ __bf16 B1l[2][BN * LDA];
    __shared__ __bf16 B2l[2][BN * LDA];
    __shared__ int tokS[BM];

    const int seg = tileSeg[blockIdx.x];
    if (seg < 0) return;                        // block-uniform
    const int e     = seg >> 1;
    const int slot0 = tileSlot[blockIdx.x];
    int rows = segOff[seg + 1] - slot0; if (rows > BM) rows = BM;
    const int n0   = blockIdx.y * BN;
    const int tid  = threadIdx.x;
    const int lane = tid & 31, wv = tid >> 5;
    const int wm = wv & 3, wn = wv >> 2;

    if (tid < BM) tokS[tid] = entryTok[slot0 + ((tid < rows) ? tid : 0)];
    __syncthreads();

    v8f accU[2][2], accV[2][2];
#pragma unroll
    for (int i = 0; i < 2; ++i)
#pragma unroll
        for (int j = 0; j < 2; ++j)
#pragma unroll
            for (int r = 0; r < 8; ++r) { accU[i][j][r] = 0.f; accV[i][j][r] = 0.f; }

    // staging roles
    const int arow = tid >> 1, ahalf = tid & 1;          // A: 2 thr/row, 32B each
    const int bgrp = tid >> 7;                           // 0 -> W1, 1 -> W2
    const int bcq  = (tid & 127) & 15;                   // col quad (4 cols)
    const int bkq  = (tid & 127) >> 4;                   // k quad (4 rows)
    const float* Wsrc = (bgrp ? W2 + (size_t)e * DD * HH : W1 + (size_t)e * DD * HH)
                        + (size_t)(bkq * 4) * HH + n0 + bcq * 4;
    const __bf16* aSrc = xb + (size_t)tokS[arow] * DD + ahalf * 16;
    const int aDst = arow * LDA + ahalf * 16;
    const unsigned aLds[2] = {(unsigned)(uintptr_t)&Al[0][aDst],
                              (unsigned)(uintptr_t)&Al[1][aDst]};

    // ---- prologue: stage kb=0 into buffer 0 ----
    {
        async_ld16(aLds[0],      aSrc);
        async_ld16(aLds[0] + 16, aSrc + 8);
        float fr[4][4];
#pragma unroll
        for (int i = 0; i < 4; ++i) {
            float4 q = *(const float4*)(Wsrc + (size_t)i * HH);
            fr[i][0] = q.x; fr[i][1] = q.y; fr[i][2] = q.z; fr[i][3] = q.w;
        }
        __bf16* Bd = bgrp ? B2l[0] : B1l[0];
#pragma unroll
        for (int j = 0; j < 4; ++j) {
            v4bf pk;
#pragma unroll
            for (int i = 0; i < 4; ++i) pk[i] = (__bf16)fr[i][j];
            *(v4bf*)&Bd[(bcq * 4 + j) * LDA + bkq * 4] = pk;
        }
    }
    wait_async0();
    __syncthreads();

    const int r16 = lane & 15;
    int p = 0;
    for (int kb = 0; kb < DD; kb += BK, p ^= 1) {
        // 1) fragment loads from buffer p
        v16bf afr[2], b1fr[2], b2fr[2];
#pragma unroll
        for (int s = 0; s < 2; ++s) {
            afr[s]  = load_fragA(Al[p],  wm * 32 + s * 16 + r16, lane);
            b1fr[s] = load_fragB(B1l[p], wn * 32 + s * 16 + r16, lane);
            b2fr[s] = load_fragB(B2l[p], wn * 32 + s * 16 + r16, lane);
        }
        // 2) kick off next tile: A via async DMA, B via registers (needs cvt)
        const bool hasNext = (kb + BK) < DD;     // uniform
        float fr[4][4];
        if (hasNext) {
            const __bf16* g = aSrc + kb + BK;
            async_ld16(aLds[p ^ 1],      g);
            async_ld16(aLds[p ^ 1] + 16, g + 8);
#pragma unroll
            for (int i = 0; i < 4; ++i) {
                float4 q = *(const float4*)(Wsrc + (size_t)(kb + BK + i) * HH);
                fr[i][0] = q.x; fr[i][1] = q.y; fr[i][2] = q.z; fr[i][3] = q.w;
            }
        }
        // 3) matrix math (overlaps the memory above)
#pragma unroll
        for (int sm = 0; sm < 2; ++sm)
#pragma unroll
            for (int sn = 0; sn < 2; ++sn) {
                accU[sm][sn] = __builtin_amdgcn_wmma_f32_16x16x32_bf16(
                    false, afr[sm], false, b1fr[sn], (short)0, accU[sm][sn], false, false);
                accV[sm][sn] = __builtin_amdgcn_wmma_f32_16x16x32_bf16(
                    false, afr[sm], false, b2fr[sn], (short)0, accV[sm][sn], false, false);
            }
        // 4) commit next B tile into buffer p^1
        if (hasNext) {
            const int pn = p ^ 1;
            __bf16* Bd = bgrp ? B2l[pn] : B1l[pn];
#pragma unroll
            for (int j = 0; j < 4; ++j) {
                v4bf pk;
#pragma unroll
                for (int i = 0; i < 4; ++i) pk[i] = (__bf16)fr[i][j];
                *(v4bf*)&Bd[(bcq * 4 + j) * LDA + bkq * 4] = pk;
            }
        }
        wait_async0();
        __syncthreads();
    }

    // ---- fused SwiGLU epilogue: h = u*sigmoid(u)*v (rcp-based) ----
    const int rowAdd = (lane < 16) ? 0 : 8;
    if (rows == BM) {                            // fast path, block-uniform
#pragma unroll
        for (int sm = 0; sm < 2; ++sm)
#pragma unroll
            for (int sn = 0; sn < 2; ++sn) {
                const int col = n0 + wn * 32 + sn * 16 + r16;
#pragma unroll
                for (int r = 0; r < 8; ++r) {
                    const int rl = wm * 32 + sm * 16 + r + rowAdd;
                    float u = accU[sm][sn][r], v = accV[sm][sn][r];
                    float h = u * __builtin_amdgcn_rcpf(1.f + __expf(-u)) * v;
                    hb[(size_t)(slot0 + rl) * HH + col] = (__bf16)h;
                }
            }
    } else {
#pragma unroll
        for (int sm = 0; sm < 2; ++sm)
#pragma unroll
            for (int sn = 0; sn < 2; ++sn) {
                const int col = n0 + wn * 32 + sn * 16 + r16;
#pragma unroll
                for (int r = 0; r < 8; ++r) {
                    const int rl = wm * 32 + sm * 16 + r + rowAdd;
                    if (rl < rows) {
                        float u = accU[sm][sn][r], v = accV[sm][sn][r];
                        float h = u * __builtin_amdgcn_rcpf(1.f + __expf(-u)) * v;
                        hb[(size_t)(slot0 + rl) * HH + col] = (__bf16)h;
                    }
                }
            }
    }
}

// ---------------------------------------------------------------------------
// Pass 5: grouped GEMM  y = h W3[e], scaled by prob; KSEL=0 writes out,
//         KSEL=1 accumulates (separate launches -> race-free, deterministic)
// ---------------------------------------------------------------------------
template<int KSEL>
__global__ __launch_bounds__(256) void moe_gemm_out(
    const __bf16* __restrict__ hb, const float* __restrict__ W3,
    float* __restrict__ out, const int* __restrict__ entryTok,
    const float* __restrict__ tokW, const int* __restrict__ tileSeg,
    const int* __restrict__ tileSlot, const int* __restrict__ segOff)
{
    __shared__ __bf16 Al[2][BM * LDA];
    __shared__ __bf16 Bl[2][BN * LDA];
    __shared__ int   tokS[BM];
    __shared__ float probS[BM];

    const int seg = tileSeg[blockIdx.x];
    if (seg < 0 || (seg & 1) != KSEL) return;   // block-uniform
    const int e     = seg >> 1;
    const int slot0 = tileSlot[blockIdx.x];
    int rows = segOff[seg + 1] - slot0; if (rows > BM) rows = BM;
    const int n0   = blockIdx.y * BN;
    const int tid  = threadIdx.x;
    const int lane = tid & 31, wv = tid >> 5;
    const int wm = wv & 3, wn = wv >> 2;

    if (tid < BM) {
        const int tk = entryTok[slot0 + ((tid < rows) ? tid : 0)];
        tokS[tid]  = tk;
        probS[tid] = tokW[tk * 2 + KSEL];
    }
    __syncthreads();

    v8f acc[2][2];
#pragma unroll
    for (int i = 0; i < 2; ++i)
#pragma unroll
        for (int j = 0; j < 2; ++j)
#pragma unroll
            for (int r = 0; r < 8; ++r) acc[i][j][r] = 0.f;

    const int arow = tid >> 1, ahalf = tid & 1;
    const int ar   = (arow < rows) ? arow : 0;   // clamp: stay inside written hb
    const int bgrp = tid >> 7;                   // only group 0 stages B
    const int bcq  = (tid & 127) & 15;
    const int bkq  = (tid & 127) >> 4;
    const float* Wsrc = W3 + (size_t)e * HH * DD + (size_t)(bkq * 4) * DD + n0 + bcq * 4;
    const __bf16* aSrc = hb + (size_t)(slot0 + ar) * HH + ahalf * 16;
    const int aDst = arow * LDA + ahalf * 16;
    const unsigned aLds[2] = {(unsigned)(uintptr_t)&Al[0][aDst],
                              (unsigned)(uintptr_t)&Al[1][aDst]};

    // ---- prologue ----
    {
        async_ld16(aLds[0],      aSrc);
        async_ld16(aLds[0] + 16, aSrc + 8);
        if (bgrp == 0) {
            float fr[4][4];
#pragma unroll
            for (int i = 0; i < 4; ++i) {
                float4 q = *(const float4*)(Wsrc + (size_t)i * DD);
                fr[i][0] = q.x; fr[i][1] = q.y; fr[i][2] = q.z; fr[i][3] = q.w;
            }
#pragma unroll
            for (int j = 0; j < 4; ++j) {
                v4bf pk;
#pragma unroll
                for (int i = 0; i < 4; ++i) pk[i] = (__bf16)fr[i][j];
                *(v4bf*)&Bl[0][(bcq * 4 + j) * LDA + bkq * 4] = pk;
            }
        }
    }
    wait_async0();
    __syncthreads();

    const int r16 = lane & 15;
    int p = 0;
    for (int kb = 0; kb < HH; kb += BK, p ^= 1) {
        v16bf afr[2], bfr[2];
#pragma unroll
        for (int s = 0; s < 2; ++s) {
            afr[s] = load_fragA(Al[p], wm * 32 + s * 16 + r16, lane);
            bfr[s] = load_fragB(Bl[p], wn * 32 + s * 16 + r16, lane);
        }
        const bool hasNext = (kb + BK) < HH;
        float fr[4][4];
        if (hasNext) {
            const __bf16* g = aSrc + kb + BK;
            async_ld16(aLds[p ^ 1],      g);
            async_ld16(aLds[p ^ 1] + 16, g + 8);
            if (bgrp == 0) {
#pragma unroll
                for (int i = 0; i < 4; ++i) {
                    float4 q = *(const float4*)(Wsrc + (size_t)(kb + BK + i) * DD);
                    fr[i][0] = q.x; fr[i][1] = q.y; fr[i][2] = q.z; fr[i][3] = q.w;
                }
            }
        }
#pragma unroll
        for (int sm = 0; sm < 2; ++sm)
#pragma unroll
            for (int sn = 0; sn < 2; ++sn)
                acc[sm][sn] = __builtin_amdgcn_wmma_f32_16x16x32_bf16(
                    false, afr[sm], false, bfr[sn], (short)0, acc[sm][sn], false, false);
        if (hasNext && bgrp == 0) {
            const int pn = p ^ 1;
#pragma unroll
            for (int j = 0; j < 4; ++j) {
                v4bf pk;
#pragma unroll
                for (int i = 0; i < 4; ++i) pk[i] = (__bf16)fr[i][j];
                *(v4bf*)&Bl[pn][(bcq * 4 + j) * LDA + bkq * 4] = pk;
            }
        }
        wait_async0();
        __syncthreads();
    }

    const int rowAdd = (lane < 16) ? 0 : 8;
    if (rows == BM) {
#pragma unroll
        for (int sm = 0; sm < 2; ++sm)
#pragma unroll
            for (int sn = 0; sn < 2; ++sn) {
                const int col = n0 + wn * 32 + sn * 16 + r16;
#pragma unroll
                for (int r = 0; r < 8; ++r) {
                    const int rl = wm * 32 + sm * 16 + r + rowAdd;
                    float* dst = out + (size_t)tokS[rl] * DD + col;
                    float val = acc[sm][sn][r] * probS[rl];
                    if (KSEL == 1) val += *dst;
                    *dst = val;
                }
            }
    } else {
#pragma unroll
        for (int sm = 0; sm < 2; ++sm)
#pragma unroll
            for (int sn = 0; sn < 2; ++sn) {
                const int col = n0 + wn * 32 + sn * 16 + r16;
#pragma unroll
                for (int r = 0; r < 8; ++r) {
                    const int rl = wm * 32 + sm * 16 + r + rowAdd;
                    if (rl < rows) {
                        float* dst = out + (size_t)tokS[rl] * DD + col;
                        float val = acc[sm][sn][r] * probS[rl];
                        if (KSEL == 1) val += *dst;
                        *dst = val;
                    }
                }
            }
    }
}

// ---------------------------------------------------------------------------
extern "C" void kernel_launch(void* const* d_in, const int* in_sizes, int n_in,
                              void* d_out, int out_size, void* d_ws, size_t ws_size,
                              hipStream_t stream)
{
    const float* x  = (const float*)d_in[0];   // [4,2048,1024]
    const float* Wg = (const float*)d_in[1];   // [1024,8]
    const float* W1 = (const float*)d_in[2];   // [8,1024,2048]
    const float* W2 = (const float*)d_in[3];   // [8,1024,2048]
    const float* W3 = (const float*)d_in[4];   // [8,2048,1024]
    float* out = (float*)d_out;                // [4,2048,1024]

    char* ws = (char*)d_ws;
    const size_t XB_BYTES = (size_t)T_TOK * DD * 2;          // 16 MB
    const size_t HB_BYTES = (size_t)(2 * T_TOK) * HH * 2;    // 64 MB
    __bf16* xb = (__bf16*)ws;
    __bf16* hb = (__bf16*)(ws + XB_BYTES);
    int*   entryTok = (int*)(ws + XB_BYTES + HB_BYTES);      // 2T ints
    int*   tokE     = entryTok + 2 * T_TOK;                  // 2T ints
    float* tokW     = (float*)(tokE + 2 * T_TOK);            // 2T floats
    int*   cnt      = (int*)(tokW + 2 * T_TOK);              // 16
    int*   fill     = cnt + NSEG;                            // 16
    int*   segOff   = fill + NSEG;                           // 17
    int*   tileSeg  = segOff + (NSEG + 1);                   // NT
    int*   tileSlot = tileSeg + NT;                          // NT

    hipMemsetAsync(cnt, 0, NSEG * sizeof(int), stream);

    moe_router<<<T_TOK / 8, 256, 0, stream>>>(x, Wg, xb, tokE, tokW, cnt);
    moe_scan<<<1, 64, 0, stream>>>(cnt, segOff, fill, tileSeg, tileSlot);
    moe_scatter<<<T_TOK / 256, 256, 0, stream>>>(tokE, fill, segOff, entryTok);
    moe_gemm_uv<<<dim3(NT, HH / BN), 256, 0, stream>>>(xb, W1, W2, hb, entryTok,
                                                       tileSeg, tileSlot, segOff);
    moe_gemm_out<0><<<dim3(NT, DD / BN), 256, 0, stream>>>(hb, W3, out, entryTok, tokW,
                                                           tileSeg, tileSlot, segOff);
    moe_gemm_out<1><<<dim3(NT, DD / BN), 256, 0, stream>>>(hb, W3, out, entryTok, tokW,
                                                           tileSeg, tileSlot, segOff);
}
